// ENCGRU_37031208026607
// MI455X (gfx1250) — matrix operand order repaired and checked by reading, not compiled, not run on hardware
//
#include <hip/hip_runtime.h>

typedef __bf16 bf16_t;
typedef __attribute__((ext_vector_type(16))) __bf16 v16bf;
typedef __attribute__((ext_vector_type(8)))  float  v8f;

#define HID    256
#define SEQ    8
#define INDIM  63
#define NNODES 21845
#define NTOK   (NNODES * SEQ)

// ---------------------------------------------------------------------------
// WMMA helpers (CDNA5 16x16x32 bf16, f32 accumulate)
// ---------------------------------------------------------------------------
__device__ __forceinline__ v8f wmma_bf16(v16bf a, v16bf b, v8f c) {
  return __builtin_amdgcn_wmma_f32_16x16x32_bf16(
      /*neg_a=*/false, a, /*neg_b=*/false, b,
      /*c_mod=*/(short)0, c, /*reuse_a=*/false, /*reuse_b=*/false);
}

// A tile 16x32 bf16 from LDS/mem, row-major, stride ld.
// Lane layout (ISA 7.12.2): lanes 0-15 -> M=lane, K = {0..7,16..23};
// lanes 16-31 -> M=lane-16, K = {8..15,24..31}.
__device__ __forceinline__ v16bf load_a_bf16(const bf16_t* base, int ld, int kbase) {
  const int lane = threadIdx.x & 31;
  const int m = lane & 15, g = lane >> 4;
  const bf16_t* p = base + m * ld + kbase + g * 8;
  v16bf a;
#pragma unroll
  for (int j = 0; j < 8; ++j) a[j] = p[j];
#pragma unroll
  for (int j = 0; j < 8; ++j) a[8 + j] = p[16 + j];
  return a;
}

__device__ __forceinline__ v16bf load_a_f32(const float* base, int ld, int kbase) {
  const int lane = threadIdx.x & 31;
  const int m = lane & 15, g = lane >> 4;
  const float* p = base + m * ld + kbase + g * 8;
  v16bf a;
#pragma unroll
  for (int j = 0; j < 8; ++j) a[j] = (bf16_t)p[j];
#pragma unroll
  for (int j = 0; j < 8; ++j) a[8 + j] = (bf16_t)p[16 + j];
  return a;
}

// A tile where all 16 rows equal vector e[K] (for M=1 head matvecs).
__device__ __forceinline__ v16bf load_a_vec(const bf16_t* e, int kbase) {
  const int g = (threadIdx.x >> 4) & 1;
  const bf16_t* p = e + kbase + g * 8;
  v16bf a;
#pragma unroll
  for (int j = 0; j < 8; ++j) a[j] = p[j];
#pragma unroll
  for (int j = 0; j < 8; ++j) a[8 + j] = p[16 + j];
  return a;
}

// B tile 32x16 for D = X @ W^T: W is (N,K) row-major bf16.
// lanes 0-15: N=lane, K=kbase+{0..7,16..23}; lanes 16-31: N=lane-16, K=kbase+{8..15,24..31}.
__device__ __forceinline__ v16bf load_b_g(const bf16_t* W, int ldk, int n0, int kbase) {
  const int lane = threadIdx.x & 31;
  const int n = lane & 15, g = lane >> 4;
  const bf16_t* p = W + (size_t)(n0 + n) * ldk + kbase + g * 8;
  v16bf b;
#pragma unroll
  for (int j = 0; j < 8; ++j) b[j] = p[j];
#pragma unroll
  for (int j = 0; j < 8; ++j) b[8 + j] = p[16 + j];
  return b;
}

// leaky relu via max/min -> v_max_num_f32 / v_min_num_f32 + fma (no cndmask)
__device__ __forceinline__ float leaky(float x) {
  return fmaxf(x, 0.f) + 0.2f * fminf(x, 0.f);
}
__device__ __forceinline__ float sigmoidf(float x) { return 1.f / (1.f + expf(-x)); }

// ---------------------------------------------------------------------------
// Weight conversion (f32 -> bf16), l1_w padded in K: 63 -> 64
// ---------------------------------------------------------------------------
__global__ __launch_bounds__(256) void convert_bf16(const float* __restrict__ src,
                                                    bf16_t* __restrict__ dst, int n) {
  int i = blockIdx.x * 256 + threadIdx.x;
  if (i < n) dst[i] = (bf16_t)src[i];
}

__global__ __launch_bounds__(256) void convert_w1(const float* __restrict__ src,
                                                  bf16_t* __restrict__ dst) {
  int i = blockIdx.x * 256 + threadIdx.x;
  if (i < 256 * 64) {
    int r = i >> 6, c = i & 63;
    dst[i] = (bf16_t)((c < INDIM) ? src[r * INDIM + c] : 0.f);
  }
}

// ---------------------------------------------------------------------------
// Fused 3-layer MLP over all tokens. WG = 256 thr (8 waves), 64 token rows.
// Waves: rowgroup = wave&3 (16 rows), N-half = wave>>2 (128 cols).
// N-tiles processed in pairs sharing one A-load -> two independent WMMA
// chains (no D->A hazard nops) and half the LDS A traffic.
// LDS 64KB: [0,8K) xin bf16 64x64; [32K,64K) h1 bf16 64x256; [0,32K) h2 (reuses xin)
// ---------------------------------------------------------------------------
__global__ __launch_bounds__(256) void mlp_kernel(
    const float* __restrict__ tokens,
    const bf16_t* __restrict__ W1, const float* __restrict__ b1,
    const bf16_t* __restrict__ W2, const float* __restrict__ b2,
    const bf16_t* __restrict__ W3, const float* __restrict__ b3,
    bf16_t* __restrict__ out) {
  __shared__ __align__(16) char smem[65536];
  bf16_t* xin = (bf16_t*)smem;             // 64 x 64
  bf16_t* h1  = (bf16_t*)(smem + 32768);   // 64 x 256
  bf16_t* h2  = (bf16_t*)smem;             // 64 x 256 (after xin is dead)

  const int tid = threadIdx.x;
  const int wave = tid >> 5;
  const int lane = tid & 31;
  const int n = lane & 15, g = lane >> 4;
  const int row0 = (wave & 3) * 16;
  const int nhalf = wave >> 2;
  const long tokbase = (long)blockIdx.x * 64;

  // stage input block as bf16, pad K 63->64
  for (int i = tid; i < 64 * 64; i += 256) {
    int r = i >> 6, c = i & 63;
    long tr = tokbase + r; if (tr >= NTOK) tr = NTOK - 1;
    xin[i] = (bf16_t)((c < INDIM) ? tokens[tr * INDIM + c] : 0.f);
  }
  // prefetch next block's token rows into cache while this block computes
  {
    long pr = tokbase + 64 + tid; if (pr >= NTOK) pr = NTOK - 1;
    __builtin_prefetch(&tokens[pr * INDIM], 0, 1);
  }
  __syncthreads();

  // ---- layer 1: K=64, leaky ----
  for (int tp = 0; tp < 4; ++tp) {
    int col0 = nhalf * 128 + tp * 32;
    float bias0 = b1[col0 + n], bias1 = b1[col0 + 16 + n];
    v8f acc0, acc1;
#pragma unroll
    for (int r = 0; r < 8; ++r) { acc0[r] = bias0; acc1[r] = bias1; }
#pragma unroll
    for (int ks = 0; ks < 2; ++ks) {
      v16bf a  = load_a_bf16(xin + row0 * 64, 64, ks * 32);
      v16bf b0 = load_b_g(W1, 64, col0, ks * 32);
      v16bf b1t = load_b_g(W1, 64, col0 + 16, ks * 32);
      acc0 = wmma_bf16(a, b0, acc0);
      acc1 = wmma_bf16(a, b1t, acc1);
    }
#pragma unroll
    for (int r = 0; r < 8; ++r) {
      int m = row0 + r + 8 * g;
      h1[m * 256 + col0 + n]      = (bf16_t)leaky(acc0[r]);
      h1[m * 256 + col0 + 16 + n] = (bf16_t)leaky(acc1[r]);
    }
  }
  __syncthreads();

  // ---- layer 2: K=256, leaky ----
  for (int tp = 0; tp < 4; ++tp) {
    int col0 = nhalf * 128 + tp * 32;
    float bias0 = b2[col0 + n], bias1 = b2[col0 + 16 + n];
    v8f acc0, acc1;
#pragma unroll
    for (int r = 0; r < 8; ++r) { acc0[r] = bias0; acc1[r] = bias1; }
    for (int ks = 0; ks < 8; ++ks) {
      v16bf a  = load_a_bf16(h1 + row0 * 256, 256, ks * 32);
      v16bf b0 = load_b_g(W2, 256, col0, ks * 32);
      v16bf b1t = load_b_g(W2, 256, col0 + 16, ks * 32);
      acc0 = wmma_bf16(a, b0, acc0);
      acc1 = wmma_bf16(a, b1t, acc1);
    }
#pragma unroll
    for (int r = 0; r < 8; ++r) {
      int m = row0 + r + 8 * g;
      h2[m * 256 + col0 + n]      = (bf16_t)leaky(acc0[r]);
      h2[m * 256 + col0 + 16 + n] = (bf16_t)leaky(acc1[r]);
    }
  }
  __syncthreads();

  // ---- layer 3: K=256, linear, store to global bf16 ----
  for (int tp = 0; tp < 4; ++tp) {
    int col0 = nhalf * 128 + tp * 32;
    float bias0 = b3[col0 + n], bias1 = b3[col0 + 16 + n];
    v8f acc0, acc1;
#pragma unroll
    for (int r = 0; r < 8; ++r) { acc0[r] = bias0; acc1[r] = bias1; }
    for (int ks = 0; ks < 8; ++ks) {
      v16bf a  = load_a_bf16(h2 + row0 * 256, 256, ks * 32);
      v16bf b0 = load_b_g(W3, 256, col0, ks * 32);
      v16bf b1t = load_b_g(W3, 256, col0 + 16, ks * 32);
      acc0 = wmma_bf16(a, b0, acc0);
      acc1 = wmma_bf16(a, b1t, acc1);
    }
#pragma unroll
    for (int r = 0; r < 8; ++r) {
      long tr = tokbase + row0 + r + 8 * g;
      if (tr < NTOK) {
        out[tr * 256 + col0 + n]      = (bf16_t)acc0[r];
        out[tr * 256 + col0 + 16 + n] = (bf16_t)acc1[r];
      }
    }
  }
}

// ---------------------------------------------------------------------------
// Fused GRU level kernel: WG = 256 thr, 16 nodes, full 8-step scan in-kernel.
// Per step: two M=16,K=256,N=768 WMMA GEMMs (x@Wih^T and h@Whh^T), gates f32.
// Waves process N-tiles in pairs; x- and h- projections use separate
// accumulators -> 4 independent WMMA chains per pair, shared A loads.
// Sequence positions 1..4 are replaced by child encodings (when not leaf).
// LDS exactly 64KB.
// ---------------------------------------------------------------------------
__global__ __launch_bounds__(256) void gru_kernel(
    const bf16_t* __restrict__ localb,   // [NTOK][256] bf16 (MLP output)
    const bf16_t* __restrict__ enc_in,   // child encodings or nullptr
    bf16_t* __restrict__ enc_out,        // [n_d][256]
    const bf16_t* __restrict__ Wih, const bf16_t* __restrict__ Whh,
    const float* __restrict__ bih, const float* __restrict__ bhh,
    const float* __restrict__ h0,
    int node_off, int n_d, int leaf) {
  __shared__ bf16_t xs[16 * 256];    //  8KB  x_t (bf16)
  __shared__ float  hs[16 * 256];    // 16KB  hidden state (f32)
  __shared__ bf16_t rzs[16 * 512];   // 16KB  r|z gates (bf16)
  __shared__ float  xns[16 * 256];   // 16KB  x-part of n gate (f32)
  __shared__ bf16_t hns[16 * 256];   //  8KB  h-part of n gate (bf16)

  const int tid = threadIdx.x;
  const int wave = tid >> 5;
  const int lane = tid & 31;
  const int n = lane & 15, g = lane >> 4;
  const int gbase = blockIdx.x * 16;

  // h0 broadcast to all 16 nodes
  for (int i = tid; i < 4096; i += 256) hs[i] = h0[i & 255];

  for (int t = 0; t < 8; ++t) {
    __syncthreads();
    // stage x_t (gather: local tokens, or child encodings for s in 1..4)
    for (int i = tid; i < 4096; i += 256) {
      int node = i >> 8, c = i & 255;
      int gn = gbase + node; if (gn >= n_d) gn = n_d - 1;
      bf16_t v;
      if (!leaf && t >= 1 && t <= 4)
        v = enc_in[((size_t)gn * 4 + (t - 1)) * 256 + c];
      else
        v = localb[((size_t)(node_off + gn) * 8 + t) * 256 + c];
      xs[i] = v;
    }
    __syncthreads();

    // phase A: 24 tile-pairs over the 768-wide gate output, 3 pairs per wave.
    // pairs 0..15 -> r,z gates (cols 0..511); pairs 16..23 -> n gate.
    for (int p = wave; p < 24; p += 8) {
      int col0 = p * 32;
      float bx0 = bih[col0 + n],      bh0 = bhh[col0 + n];
      float bx1 = bih[col0 + 16 + n], bh1 = bhh[col0 + 16 + n];
      v8f accx0, acch0, accx1, acch1;
#pragma unroll
      for (int r = 0; r < 8; ++r) {
        accx0[r] = bx0; acch0[r] = bh0; accx1[r] = bx1; acch1[r] = bh1;
      }
      for (int ks = 0; ks < 8; ++ks) {
        v16bf a_x = load_a_bf16(xs, 256, ks * 32);
        v16bf a_h = load_a_f32(hs, 256, ks * 32);
        v16bf bi0 = load_b_g(Wih, 256, col0,      ks * 32);
        v16bf bi1 = load_b_g(Wih, 256, col0 + 16, ks * 32);
        v16bf bh0v = load_b_g(Whh, 256, col0,      ks * 32);
        v16bf bh1v = load_b_g(Whh, 256, col0 + 16, ks * 32);
        accx0 = wmma_bf16(a_x, bi0,  accx0);
        accx1 = wmma_bf16(a_x, bi1,  accx1);
        acch0 = wmma_bf16(a_h, bh0v, acch0);
        acch1 = wmma_bf16(a_h, bh1v, acch1);
      }
      if (p < 16) {  // r,z: combined projection through sigmoid
#pragma unroll
        for (int r = 0; r < 8; ++r) {
          int m = r + 8 * g;
          rzs[m * 512 + col0 + n]      = (bf16_t)sigmoidf(accx0[r] + acch0[r]);
          rzs[m * 512 + col0 + 16 + n] = (bf16_t)sigmoidf(accx1[r] + acch1[r]);
        }
      } else {       // n gate: keep x- and h- parts separate
        int cn = col0 - 512 + n;
#pragma unroll
        for (int r = 0; r < 8; ++r) {
          int m = r + 8 * g;
          xns[m * 256 + cn]      = accx0[r];
          hns[m * 256 + cn]      = (bf16_t)acch0[r];
          xns[m * 256 + cn + 16] = accx1[r];
          hns[m * 256 + cn + 16] = (bf16_t)acch1[r];
        }
      }
    }
    __syncthreads();

    // phase B: gate elementwise + state update (f32)
    for (int i = tid; i < 4096; i += 256) {
      int node = i >> 8, c = i & 255;
      float r = (float)rzs[node * 512 + c];
      float z = (float)rzs[node * 512 + 256 + c];
      float nn = tanhf(xns[i] + r * (float)hns[i]);
      hs[i] = (1.f - z) * nn + z * hs[i];
    }
  }
  __syncthreads();

  for (int i = tid; i < 4096; i += 256) {
    int node = i >> 8, c = i & 255;
    int gn = gbase + node;
    if (gn < n_d) enc_out[(size_t)gn * 256 + c] = (bf16_t)hs[i];
  }
}

// ---------------------------------------------------------------------------
// Head: e = relu(root @ s1^T + b); mu = e @ smu^T + b. Single WG; rows of the
// A tile are all the same vector so every D row is identical; row 0 stored.
// ---------------------------------------------------------------------------
__global__ __launch_bounds__(256) void head_kernel(
    const bf16_t* __restrict__ enc_root,
    const bf16_t* __restrict__ Ws1, const float* __restrict__ bs1,
    const bf16_t* __restrict__ Wsmu, const float* __restrict__ bsmu,
    float* __restrict__ out) {
  __shared__ bf16_t root[256];
  __shared__ bf16_t ev[256];
  const int tid = threadIdx.x;
  const int wave = tid >> 5;
  const int lane = tid & 31;
  const int n = lane & 15, g = lane >> 4;

  if (tid < 256) root[tid] = enc_root[tid];
  __syncthreads();

  {  // layer s1: each wave owns a pair of adjacent N-tiles
    int col0 = wave * 32;
    float bias0 = bs1[col0 + n], bias1 = bs1[col0 + 16 + n];
    v8f acc0, acc1;
#pragma unroll
    for (int r = 0; r < 8; ++r) { acc0[r] = bias0; acc1[r] = bias1; }
    for (int ks = 0; ks < 8; ++ks) {
      v16bf a  = load_a_vec(root, ks * 32);
      v16bf b0 = load_b_g(Ws1, 256, col0, ks * 32);
      v16bf b1t = load_b_g(Ws1, 256, col0 + 16, ks * 32);
      acc0 = wmma_bf16(a, b0, acc0);
      acc1 = wmma_bf16(a, b1t, acc1);
    }
    if (g == 0) {
      ev[col0 + n]      = (bf16_t)fmaxf(acc0[0], 0.f);
      ev[col0 + 16 + n] = (bf16_t)fmaxf(acc1[0], 0.f);
    }
  }
  __syncthreads();

  {  // layer smu
    int col0 = wave * 32;
    float bias0 = bsmu[col0 + n], bias1 = bsmu[col0 + 16 + n];
    v8f acc0, acc1;
#pragma unroll
    for (int r = 0; r < 8; ++r) { acc0[r] = bias0; acc1[r] = bias1; }
    for (int ks = 0; ks < 8; ++ks) {
      v16bf a  = load_a_vec(ev, ks * 32);
      v16bf b0 = load_b_g(Wsmu, 256, col0, ks * 32);
      v16bf b1t = load_b_g(Wsmu, 256, col0 + 16, ks * 32);
      acc0 = wmma_bf16(a, b0, acc0);
      acc1 = wmma_bf16(a, b1t, acc1);
    }
    if (g == 0) {
      out[col0 + n]      = acc0[0];
      out[col0 + 16 + n] = acc1[0];
    }
  }
}

// ---------------------------------------------------------------------------
extern "C" void kernel_launch(void* const* d_in, const int* in_sizes, int n_in,
                              void* d_out, int out_size, void* d_ws, size_t ws_size,
                              hipStream_t stream) {
  const float* tokens  = (const float*)d_in[0];
  const float* l1_w    = (const float*)d_in[1];
  const float* l1_b    = (const float*)d_in[2];
  const float* l2_w    = (const float*)d_in[3];
  const float* l2_b    = (const float*)d_in[4];
  const float* l3_w    = (const float*)d_in[5];
  const float* l3_b    = (const float*)d_in[6];
  const float* gru_wih = (const float*)d_in[7];
  const float* gru_whh = (const float*)d_in[8];
  const float* gru_bih = (const float*)d_in[9];
  const float* gru_bhh = (const float*)d_in[10];
  const float* h_start = (const float*)d_in[11];
  const float* s1_w    = (const float*)d_in[12];
  const float* s1_b    = (const float*)d_in[13];
  const float* smu_w   = (const float*)d_in[14];
  const float* smu_b   = (const float*)d_in[15];

  char* ws = (char*)d_ws;
  size_t off = 0;
  auto alloc = [&](size_t bytes) -> char* {
    char* p = ws + off;
    off = (off + bytes + 255) & ~(size_t)255;
    return p;
  };
  bf16_t* W1     = (bf16_t*)alloc(256 * 64 * 2);
  bf16_t* W2     = (bf16_t*)alloc(256 * 256 * 2);
  bf16_t* W3     = (bf16_t*)alloc(256 * 256 * 2);
  bf16_t* Wih    = (bf16_t*)alloc(768 * 256 * 2);
  bf16_t* Whh    = (bf16_t*)alloc(768 * 256 * 2);
  bf16_t* Ws1    = (bf16_t*)alloc(256 * 256 * 2);
  bf16_t* Wsmu   = (bf16_t*)alloc(256 * 256 * 2);
  bf16_t* localb = (bf16_t*)alloc((size_t)NTOK * 256 * 2);
  bf16_t* encA   = (bf16_t*)alloc((size_t)16384 * 256 * 2);
  bf16_t* encB   = (bf16_t*)alloc((size_t)16384 * 256 * 2);

  // weight conversion to bf16
  convert_w1<<<(256 * 64 + 255) / 256, 256, 0, stream>>>(l1_w, W1);
  convert_bf16<<<(65536 + 255) / 256, 256, 0, stream>>>(l2_w, W2, 65536);
  convert_bf16<<<(65536 + 255) / 256, 256, 0, stream>>>(l3_w, W3, 65536);
  convert_bf16<<<(196608 + 255) / 256, 256, 0, stream>>>(gru_wih, Wih, 196608);
  convert_bf16<<<(196608 + 255) / 256, 256, 0, stream>>>(gru_whh, Whh, 196608);
  convert_bf16<<<(65536 + 255) / 256, 256, 0, stream>>>(s1_w, Ws1, 65536);
  convert_bf16<<<(65536 + 255) / 256, 256, 0, stream>>>(smu_w, Wsmu, 65536);

  // token MLP (all nodes, all sequence positions)
  mlp_kernel<<<(NTOK + 63) / 64, 256, 0, stream>>>(tokens, W1, l1_b, W2, l2_b,
                                                   W3, l3_b, localb);

  // bottom-up tree GRU, one launch per level (serial dependence between levels)
  bf16_t* bufs[2] = {encA, encB};
  const bf16_t* enc_in = nullptr;
  for (int d = 7; d >= 0; --d) {
    int n_d = 1 << (2 * d);
    int node_off = ((1 << (2 * d)) - 1) / 3;
    bf16_t* enc_out = bufs[d & 1];
    gru_kernel<<<(n_d + 15) / 16, 256, 0, stream>>>(
        localb, enc_in, enc_out, Wih, Whh, gru_bih, gru_bhh, h_start,
        node_off, n_d, (d == 7) ? 1 : 0);
    enc_in = enc_out;
  }

  // projection head on the root encoding
  head_kernel<<<1, 256, 0, stream>>>(bufs[0], Ws1, s1_b, Wsmu, smu_b,
                                     (float*)d_out);
}
